// Attention_3607772528689
// MI455X (gfx1250) — matrix operand order repaired and checked by reading, not compiled, not run on hardware
//
#include <hip/hip_runtime.h>

// ---------------- problem constants ----------------
#define DIMC    256
#define HEADS   8
#define BATCH   8
#define SEQ     2048
#define M_TOTAL (BATCH * SEQ)        // 16384 rows
#define QKV_N   (3 * HEADS * DIMC)   // 6144

// ---------------- plain vector types (union-safe, no HIP ctors) --------
typedef __attribute__((ext_vector_type(16))) __bf16        bf16x16;
typedef __attribute__((ext_vector_type(8)))  float         f32x8;
typedef __attribute__((ext_vector_type(4)))  float         f32x4;
typedef __attribute__((ext_vector_type(4)))  unsigned int  u32x4;
typedef __attribute__((ext_vector_type(2)))  unsigned int  u32x2;
typedef int v4i __attribute__((__vector_size__(16)));      // matches builtin proto

// ---------------- CDNA5 async global->LDS copy (ASYNCcnt path) ---------
#if defined(__has_builtin)
# if __has_builtin(__builtin_amdgcn_global_load_async_to_lds_b128) && \
     __has_builtin(__builtin_amdgcn_s_wait_asynccnt)
#  define USE_ASYNC_LDS 1
# endif
#endif
#ifndef USE_ASYNC_LDS
# define USE_ASYNC_LDS 0
#endif

static __device__ __forceinline__ void cp16_async(unsigned short* l,
                                                  const unsigned short* g) {
#if USE_ASYNC_LDS
    __builtin_amdgcn_global_load_async_to_lds_b128(
        (__attribute__((address_space(1))) v4i*)(g),
        (__attribute__((address_space(3))) v4i*)(l), 0, 0);
#else
    *(u32x4*)l = *(const u32x4*)g;   // sync fallback: vmem load + ds_store
#endif
}
static __device__ __forceinline__ void wait_async0() {
#if USE_ASYNC_LDS
    __builtin_amdgcn_s_wait_asynccnt(0);
#endif
}

static __device__ __forceinline__ unsigned short f32_bf16(float f) {
    unsigned int u = __float_as_uint(f);
    u += 0x7FFFu + ((u >> 16) & 1u);      // round-to-nearest-even
    return (unsigned short)(u >> 16);
}

static __device__ __forceinline__ f32x8 wmma_bf16(bf16x16 a, bf16x16 b, f32x8 c) {
    return __builtin_amdgcn_wmma_f32_16x16x32_bf16(false, a, false, b,
                                                   (short)0, c, false, false);
}

// A fragment (16x32 bf16, row-major, leading dim `ld`):
//   lanes 0-15 : M=lane,    K=[k0..k0+8) and [k0+16..k0+24)
//   lanes16-31 : M=lane-16, K=[k0+8..k0+16) and [k0+24..k0+32)
static __device__ __forceinline__ bf16x16
frag_a(const unsigned short* p, int row0, int ld, int k0, int lane) {
    int m  = row0 + (lane & 15);
    int kk = k0 + ((lane >> 4) << 3);
    union { u32x4 u[2]; bf16x16 v; } f;
    f.u[0] = *(const u32x4*)(p + m * ld + kk);
    f.u[1] = *(const u32x4*)(p + m * ld + kk + 16);
    return f.v;
}

// B fragment (32x16). Source rows hold K-contiguous data for one output col:
//   lanes 0-15 : N=lane,    K=[k0..k0+8) and [k0+16..k0+24)
//   lanes16-31 : N=lane-16, K=[k0+8..k0+16) and [k0+24..k0+32)
static __device__ __forceinline__ bf16x16
frag_b(const unsigned short* p, int col0, int ld, int k0, int lane) {
    int n  = col0 + (lane & 15);
    int kk = k0 + ((lane >> 4) << 3);
    union { u32x4 u[2]; bf16x16 v; } f;
    f.u[0] = *(const u32x4*)(p + n * ld + kk);
    f.u[1] = *(const u32x4*)(p + n * ld + kk + 16);
    return f.v;
}

// ======================= 1. LayerNorm -> bf16 =========================
__global__ __launch_bounds__(256) void ln_kernel(const float* __restrict__ x,
                                                 const float* __restrict__ gamma,
                                                 const float* __restrict__ beta,
                                                 unsigned short* __restrict__ xn) {
    int row  = blockIdx.x * 8 + (threadIdx.x >> 5);
    int lane = threadIdx.x & 31;
    const float* xr = x + (size_t)row * DIMC + lane * 8;
    f32x4 a = *(const f32x4*)xr;
    f32x4 b = *(const f32x4*)(xr + 4);
    float v[8] = {a.x, a.y, a.z, a.w, b.x, b.y, b.z, b.w};

    float s = 0.f;
    for (int i = 0; i < 8; ++i) s += v[i];
    for (int m = 16; m >= 1; m >>= 1) s += __shfl_xor(s, m, 32);
    float mu = s * (1.0f / 256.0f);

    float q = 0.f;
    for (int i = 0; i < 8; ++i) { float d = v[i] - mu; q += d * d; }
    for (int m = 16; m >= 1; m >>= 1) q += __shfl_xor(q, m, 32);
    float rstd = rsqrtf(q * (1.0f / 256.0f) + 1e-5f);

    f32x4 g0 = *(const f32x4*)(gamma + lane * 8);
    f32x4 g1 = *(const f32x4*)(gamma + lane * 8 + 4);
    f32x4 b0 = *(const f32x4*)(beta + lane * 8);
    f32x4 b1 = *(const f32x4*)(beta + lane * 8 + 4);
    float g[8]  = {g0.x, g0.y, g0.z, g0.w, g1.x, g1.y, g1.z, g1.w};
    float be[8] = {b0.x, b0.y, b0.z, b0.w, b1.x, b1.y, b1.z, b1.w};

    union { unsigned short s16[8]; u32x4 u; } pk;
    for (int i = 0; i < 8; ++i)
        pk.s16[i] = f32_bf16((v[i] - mu) * rstd * g[i] + be[i]);
    *(u32x4*)(xn + (size_t)row * DIMC + lane * 8) = pk.u;
}

// ======================= 2. weights f32 -> bf16 =======================
__global__ __launch_bounds__(256) void cvt_kernel(const float* __restrict__ wqkv,
                                                  const float* __restrict__ wout,
                                                  unsigned short* __restrict__ wqb,
                                                  unsigned short* __restrict__ wob) {
    int i = blockIdx.x * 256 + threadIdx.x;     // float4 index
    const int N1 = (QKV_N * DIMC) / 4;          // 393216
    f32x4 v;
    unsigned short* dst;
    if (i < N1) { v = *(const f32x4*)(wqkv + (size_t)i * 4); dst = wqb + (size_t)i * 4; }
    else {
        int j = i - N1;                          // < 131072
        v = *(const f32x4*)(wout + (size_t)j * 4); dst = wob + (size_t)j * 4;
    }
    union { unsigned short s16[4]; u32x2 u; } pk;
    pk.s16[0] = f32_bf16(v.x); pk.s16[1] = f32_bf16(v.y);
    pk.s16[2] = f32_bf16(v.z); pk.s16[3] = f32_bf16(v.w);
    *(u32x2*)dst = pk.u;
}

// ======================= 3. QKV GEMM ==================================
// C[m,o] = sum_k xn[m,k]*wqkv[o,k]; bf16 result scattered into [3][b][h][n][d].
// Ping-pong async-staged A/B tiles; epilogue reuses the same 32 KB of LDS.
__global__ __launch_bounds__(256) void qkv_gemm(const unsigned short* __restrict__ xn,
                                                const unsigned short* __restrict__ wq,
                                                unsigned short* __restrict__ qkv) {
    __shared__ __align__(16) unsigned short smem[16384];   // 32 KB: [A0|A1|B0|B1]

    int m0 = blockIdx.x * 128, o0 = blockIdx.y * 128;
    int tid = threadIdx.x, lane = tid & 31, wv = tid >> 5;
    int wm = (wv >> 2) * 64, wn = (wv & 3) * 32;

    f32x8 zero = {0.f, 0.f, 0.f, 0.f, 0.f, 0.f, 0.f, 0.f};
    f32x8 acc[4][2];
    for (int mt = 0; mt < 4; ++mt) for (int nt = 0; nt < 2; ++nt) acc[mt][nt] = zero;

    int srow = tid >> 1, sc = (tid & 1) * 16;
    int soff = srow * 32 + sc;
    const unsigned short* ga0 = xn + (size_t)(m0 + srow) * DIMC + sc;
    const unsigned short* gb0 = wq + (size_t)(o0 + srow) * DIMC + sc;

    // prologue: stage K-step 0 into buffer 0
    cp16_async(smem + soff,            ga0);
    cp16_async(smem + soff + 8,        ga0 + 8);
    cp16_async(smem + 8192 + soff,     gb0);
    cp16_async(smem + 8192 + soff + 8, gb0 + 8);

    for (int kt = 0, it = 0; kt < DIMC; kt += 32, ++it) {
        int p = it & 1;
        unsigned short* lap = smem + p * 4096;
        unsigned short* lbp = smem + 8192 + p * 4096;
        wait_async0();
        __syncthreads();
        if (kt + 32 < DIMC) {                      // stage next tile into p^1
            unsigned short* lan = smem + (p ^ 1) * 4096;
            unsigned short* lbn = smem + 8192 + (p ^ 1) * 4096;
            cp16_async(lan + soff,     ga0 + kt + 32);
            cp16_async(lan + soff + 8, ga0 + kt + 40);
            cp16_async(lbn + soff,     gb0 + kt + 32);
            cp16_async(lbn + soff + 8, gb0 + kt + 40);
        }
        bf16x16 bfr[2];
        for (int nt = 0; nt < 2; ++nt) bfr[nt] = frag_b(lbp, wn + nt * 16, 32, 0, lane);
        for (int mt = 0; mt < 4; ++mt) {
            bf16x16 af = frag_a(lap, wm + mt * 16, 32, 0, lane);
            for (int nt = 0; nt < 2; ++nt)
                acc[mt][nt] = wmma_bf16(af, bfr[nt], acc[mt][nt]);
        }
    }
    __syncthreads();
    // epilogue: accum -> bf16 tile in smem (reused as [128][128])
    for (int mt = 0; mt < 4; ++mt) {
        int mb = wm + mt * 16 + ((lane >> 4) << 3);
        for (int nt = 0; nt < 2; ++nt) {
            int nc = wn + nt * 16 + (lane & 15);
            for (int r = 0; r < 8; ++r)
                smem[(mb + r) * 128 + nc] = f32_bf16(acc[mt][nt][r]);
        }
    }
    __syncthreads();
    // scatter: a 128-wide o0 block lies in exactly one (c,h); rows contiguous in d
    int c_ = o0 >> 11, h = (o0 >> 8) & 7, d0 = o0 & 255;
    int b_ = m0 >> 11, n0 = m0 & (SEQ - 1);
    unsigned short* dst = qkv + (size_t)(c_ * 64 + b_ * 8 + h) * (SEQ * DIMC)
                              + (size_t)n0 * DIMC + d0;
    for (int it = 0; it < 8; ++it) {
        int u4i = tid + it * 256;                 // 2048 uint4 = 128 rows * 16
        int row = u4i >> 4, cc = (u4i & 15) << 3;
        *(u32x4*)(dst + (size_t)row * DIMC + cc) = *(const u32x4*)(smem + row * 128 + cc);
    }
}

// ======================= 4. flash attention ===========================
// one block per (b, h, 64-row q tile); Q fragments live in registers,
// K tiles double-buffered via async-to-LDS, V^T overwrites the consumed K tile.
__global__ __launch_bounds__(256) void attn_kernel(const unsigned short* __restrict__ qkv,
                                                   unsigned short* __restrict__ aout) {
    __shared__ __align__(16) unsigned short lds_kv[2 * 8192];  // 32 KB (K ping-pong / V^T / O out)
    __shared__ __align__(16) float          lds_s[64 * 32];    // 8 KB raw scores
    __shared__ __align__(16) unsigned short lds_p[64 * 32];    // 4 KB probs (bf16)
    __shared__ float row_max[64], row_sum[64], row_scale[64];

    int blk = blockIdx.x;
    int qt = blk & 31, h = (blk >> 5) & 7, b = blk >> 8;
    int tid = threadIdx.x, lane = tid & 31, wv = tid >> 5;
    int smi = wv >> 1, sni = wv & 1;                 // this wave's S tile

    const size_t bh = (size_t)(b * 8 + h) * (SEQ * DIMC);
    const unsigned short* Qg = qkv + bh + (size_t)qt * 64 * DIMC;
    const unsigned short* Kg = qkv + (size_t)64  * (SEQ * DIMC) + bh;
    const unsigned short* Vg = qkv + (size_t)128 * (SEQ * DIMC) + bh;

    // ---- preload this wave's 8 Q A-fragments into registers (64 VGPRs) ----
    bf16x16 qfrag[8];
    {
        const unsigned short* qr = Qg + (size_t)(smi * 16 + (lane & 15)) * DIMC
                                      + ((lane >> 4) << 3);
        for (int k8 = 0; k8 < 8; ++k8) {
            union { u32x4 u[2]; bf16x16 v; } f;
            f.u[0] = *(const u32x4*)(qr + k8 * 32);
            f.u[1] = *(const u32x4*)(qr + k8 * 32 + 16);
            qfrag[k8] = f.v;
        }
    }
    if (tid < 64) { row_max[tid] = -3.0e38f; row_sum[tid] = 0.f; }

    f32x8 zero = {0.f, 0.f, 0.f, 0.f, 0.f, 0.f, 0.f, 0.f};
    f32x8 o_acc[4][2];                               // 64 rows x 32 cols (cols wv*32..)
    for (int mt = 0; mt < 4; ++mt) for (int nt = 0; nt < 2; ++nt) o_acc[mt][nt] = zero;

    // prologue: async-stage K tile 0
    for (int it = 0; it < 4; ++it) {
        int u4i = tid + it * 256;
        cp16_async(lds_kv + u4i * 8, Kg + (size_t)u4i * 8);
    }

    for (int kt = 0; kt < SEQ / 32; ++kt) {
        int p = kt & 1;
        unsigned short* kvp = lds_kv + p * 8192;
        wait_async0();
        __syncthreads();                             // (A) K[kt] visible, kv[p^1] free
        if (kt + 1 < SEQ / 32) {                     // async-stage K[kt+1]
            const unsigned short* Kt = Kg + (size_t)(kt + 1) * 32 * DIMC;
            unsigned short* kvn = lds_kv + (p ^ 1) * 8192;
            for (int it = 0; it < 4; ++it) {
                int u4i = tid + it * 256;
                cp16_async(kvn + u4i * 8, Kt + (size_t)u4i * 8);
            }
        }
        // hoist V tile loads (latency hidden behind S-phase WMMAs)
        const unsigned short* Vt = Vg + (size_t)kt * 32 * DIMC;
        u32x4 vbuf[4];
        for (int it = 0; it < 4; ++it) {
            int task = tid + it * 256;
            vbuf[it] = *(const u32x4*)(Vt + (size_t)(task >> 5) * DIMC + ((task & 31) << 3));
        }
        // ---- S = Q * K^T : one 16x16 tile per wave, K-dim 256 ----
        f32x8 s = zero;
        for (int k8 = 0; k8 < 8; ++k8) {
            bf16x16 bf = frag_b(kvp, sni * 16, 256, k8 * 32, lane);
            s = wmma_bf16(qfrag[k8], bf, s);
        }
        {
            int sm = smi * 16 + ((lane >> 4) << 3);
            int sn = sni * 16 + (lane & 15);
            for (int r = 0; r < 8; ++r) lds_s[(sm + r) * 32 + sn] = s[r];
        }
        __syncthreads();                             // (B) S visible; K reads done
        // scatter V^T into kvp (overwrites consumed K tile)
        for (int it = 0; it < 4; ++it) {
            int task = tid + it * 256;
            int vr = task >> 5, d0 = (task & 31) << 3;
            union { u32x4 u; unsigned short s16[8]; } dd; dd.u = vbuf[it];
            for (int j = 0; j < 8; ++j) kvp[(d0 + j) * 32 + vr] = dd.s16[j];
        }
        // online softmax: one thread per row
        if (tid < 64) {
            int r = tid;
            float om = row_max[r], mn = om;
            for (int c = 0; c < 32; ++c) mn = fmaxf(mn, lds_s[r * 32 + c]);
            float sc = __expf(om - mn);
            float sum = row_sum[r] * sc;
            for (int c = 0; c < 32; ++c) {
                float pv = __expf(lds_s[r * 32 + c] - mn);
                sum += pv;
                lds_p[r * 32 + c] = f32_bf16(pv);
            }
            row_max[r] = mn; row_sum[r] = sum; row_scale[r] = sc;
        }
        __syncthreads();                             // (C) V^T + P + scales visible
        // rescale O, then O += P * V
        for (int mt = 0; mt < 4; ++mt) {
            int mb = mt * 16 + ((lane >> 4) << 3);
            for (int r = 0; r < 8; ++r) {
                float sc = row_scale[mb + r];
                o_acc[mt][0][r] *= sc;
                o_acc[mt][1][r] *= sc;
            }
        }
        bf16x16 vb[2];
        for (int nt = 0; nt < 2; ++nt)
            vb[nt] = frag_b(kvp, wv * 32 + nt * 16, 32, 0, lane);   // V^T rows = d cols
        for (int mt = 0; mt < 4; ++mt) {
            bf16x16 pa = frag_a(lds_p, mt * 16, 32, 0, lane);
            for (int nt = 0; nt < 2; ++nt)
                o_acc[mt][nt] = wmma_bf16(pa, vb[nt], o_acc[mt][nt]);
        }
        // no trailing barrier: next iteration's (A) orders buffer reuse
    }
    __syncthreads();
    // ---- normalize, stage O (bf16) into lds_kv as [64][256], store ----
    for (int mt = 0; mt < 4; ++mt) {
        int mb = mt * 16 + ((lane >> 4) << 3);
        for (int nt = 0; nt < 2; ++nt) {
            int col = wv * 32 + nt * 16 + (lane & 15);
            for (int r = 0; r < 8; ++r) {
                float v = o_acc[mt][nt][r] / row_sum[mb + r];
                lds_kv[(mb + r) * 256 + col] = f32_bf16(v);
            }
        }
    }
    __syncthreads();
    unsigned short* Og = aout + (size_t)(b * SEQ + qt * 64) * (HEADS * DIMC) + h * DIMC;
    for (int it = 0; it < 8; ++it) {
        int u4i = tid + it * 256;                    // 2048 uint4
        int row = u4i >> 5, cc = (u4i & 31) << 3;
        *(u32x4*)(Og + (size_t)row * (HEADS * DIMC) + cc) = *(const u32x4*)(lds_kv + row * 256 + cc);
    }
}

// ======================= 5. output projection =========================
// out[m,j] = sum_o attn[m,o]*wout[j,o]; M=16384, N=256, K=2048, f32 out
__global__ __launch_bounds__(256) void out_gemm(const unsigned short* __restrict__ ain,
                                                const unsigned short* __restrict__ wo,
                                                float* __restrict__ out) {
    __shared__ __align__(16) unsigned short smem[16384];   // 32 KB: [A0|A1|B0|B1]

    int m0 = blockIdx.x * 128, n0 = blockIdx.y * 128;
    int tid = threadIdx.x, lane = tid & 31, wv = tid >> 5;
    int wm = (wv >> 2) * 64, wn = (wv & 3) * 32;
    const int K = HEADS * DIMC;                      // 2048

    f32x8 zero = {0.f, 0.f, 0.f, 0.f, 0.f, 0.f, 0.f, 0.f};
    f32x8 acc[4][2];
    for (int mt = 0; mt < 4; ++mt) for (int nt = 0; nt < 2; ++nt) acc[mt][nt] = zero;

    int srow = tid >> 1, sc = (tid & 1) * 16;
    int soff = srow * 32 + sc;
    const unsigned short* ga0 = ain + (size_t)(m0 + srow) * K + sc;
    const unsigned short* gb0 = wo  + (size_t)(n0 + srow) * K + sc;

    cp16_async(smem + soff,            ga0);
    cp16_async(smem + soff + 8,        ga0 + 8);
    cp16_async(smem + 8192 + soff,     gb0);
    cp16_async(smem + 8192 + soff + 8, gb0 + 8);

    for (int kt = 0, it = 0; kt < K; kt += 32, ++it) {
        int p = it & 1;
        unsigned short* lap = smem + p * 4096;
        unsigned short* lbp = smem + 8192 + p * 4096;
        wait_async0();
        __syncthreads();
        if (kt + 32 < K) {
            unsigned short* lan = smem + (p ^ 1) * 4096;
            unsigned short* lbn = smem + 8192 + (p ^ 1) * 4096;
            cp16_async(lan + soff,     ga0 + kt + 32);
            cp16_async(lan + soff + 8, ga0 + kt + 40);
            cp16_async(lbn + soff,     gb0 + kt + 32);
            cp16_async(lbn + soff + 8, gb0 + kt + 40);
        }
        bf16x16 bfr[2];
        for (int nt = 0; nt < 2; ++nt) bfr[nt] = frag_b(lbp, wn + nt * 16, 32, 0, lane);
        for (int mt = 0; mt < 4; ++mt) {
            bf16x16 af = frag_a(lap, wm + mt * 16, 32, 0, lane);
            for (int nt = 0; nt < 2; ++nt)
                acc[mt][nt] = wmma_bf16(af, bfr[nt], acc[mt][nt]);
        }
    }
    for (int mt = 0; mt < 4; ++mt) {
        int mb = m0 + wm + mt * 16 + ((lane >> 4) << 3);
        for (int nt = 0; nt < 2; ++nt) {
            int nc = n0 + wn + nt * 16 + (lane & 15);
            for (int r = 0; r < 8; ++r)
                out[(size_t)(mb + r) * DIMC + nc] = acc[mt][nt][r];
        }
    }
}

// ======================= host launcher ================================
extern "C" void kernel_launch(void* const* d_in, const int* in_sizes, int n_in,
                              void* d_out, int out_size, void* d_ws, size_t ws_size,
                              hipStream_t stream) {
    const float* x     = (const float*)d_in[0];
    const float* w_qkv = (const float*)d_in[1];
    const float* w_out = (const float*)d_in[2];
    const float* gamma = (const float*)d_in[3];
    const float* beta  = (const float*)d_in[4];
    float* out = (float*)d_out;

    // workspace layout (bytes), total ~268 MB:
    //   xn bf16   @ 0            (8,388,608)
    //   wqkv bf16 @ 8,388,608    (3,145,728)
    //   wout bf16 @ 11,534,336   (1,048,576)
    //   qkv bf16  @ 12,582,912   (201,326,592)  [3][b][h][n][d]
    //   attn bf16 @ 213,909,504  (67,108,864)   [m][2048]
    char* ws = (char*)d_ws;
    unsigned short* xn  = (unsigned short*)(ws);
    unsigned short* wqb = (unsigned short*)(ws + 8388608);
    unsigned short* wob = (unsigned short*)(ws + 11534336);
    unsigned short* qkv = (unsigned short*)(ws + 12582912);
    unsigned short* ao  = (unsigned short*)(ws + 213909504);

    ln_kernel  <<<M_TOTAL / 8, 256, 0, stream>>>(x, gamma, beta, xn);
    cvt_kernel <<<2048, 256, 0, stream>>>(w_qkv, w_out, wqb, wob);
    qkv_gemm   <<<dim3(M_TOTAL / 128, QKV_N / 128), 256, 0, stream>>>(xn, wqb, qkv);
    attn_kernel<<<BATCH * HEADS * (SEQ / 64), 256, 0, stream>>>(qkv, ao);
    out_gemm   <<<dim3(M_TOTAL / 128, DIMC / 128), 256, 0, stream>>>(ao, wob, out);
}
